// Attention_25039659335898
// MI455X (gfx1250) — matrix operand order repaired and checked by reading, not compiled, not run on hardware
//
#include <hip/hip_runtime.h>
#include <hip/hip_bf16.h>

typedef __attribute__((ext_vector_type(16))) _Float16 v16h;
typedef __attribute__((ext_vector_type(8)))  _Float16 v8h;
typedef __attribute__((ext_vector_type(8)))  float    v8f;

constexpr int HEADS = 8;
constexpr int BATCH = 8;
constexpr int NP    = 1024;   // sequence length
constexpr int DM    = 512;    // model dim
constexpr int DHD   = 64;     // head dim

// ---------------------------------------------------------------- conversions
__global__ __launch_bounds__(256) void cvt_f32_to_f16(const float* __restrict__ s,
                                                      _Float16* __restrict__ d, int n) {
    int i = blockIdx.x * 256 + threadIdx.x;
    if (i < n) d[i] = (_Float16)s[i];
}

// 32x32-tiled transpose with optional f32->f16 convert: dst[c*rows + r] = src[r*cols + c]
// block = (32,8), grid = (cols/32, rows/32, batches)
template <typename SrcT>
__global__ __launch_bounds__(256) void transpose_cvt_kernel(const SrcT* __restrict__ src,
        _Float16* __restrict__ dst, int rows, int cols) {
    __shared__ float tile[32][33];
    const int tx = threadIdx.x, ty = threadIdx.y;
    const size_t bofs = (size_t)blockIdx.z * rows * cols;
    const int c  = blockIdx.x * 32 + tx;
    const int r0 = blockIdx.y * 32;
    for (int j = ty; j < 32; j += 8)
        tile[j][tx] = (float)src[bofs + (size_t)(r0 + j) * cols + c];
    __syncthreads();
    const int rr = r0 + tx;
    for (int j = ty; j < 32; j += 8)
        dst[bofs + (size_t)(blockIdx.x * 32 + j) * rows + rr] = (_Float16)tile[tx][j];
}

// -------------------------------------------------------- fragment load utils
// A-matrix 16x32 f16: lane m = lane%16 holds K = {0..15}
// (+16 for lanes 16..31). p must point at rowBase + k0 + (laneHi?16:0).
__device__ __forceinline__ v16h load_a_contig(const _Float16* p) {
    return *reinterpret_cast<const v16h*>(p);
}

// B-matrix 32x16 f16 from K-contiguous (pre-transposed) storage:
// element i -> K = k0 + i + (laneHi?16:0); one 32B vector load.
__device__ __forceinline__ v16h load_b_contig(const _Float16* p) {
    return *reinterpret_cast<const v16h*>(p);
}

// ------------------------------------------------------------- 512-K GEMM
// X[M,512] f16 @ Wt[512(n),512(k)] f16 + bias -> out32 / out16 (optional each).
// grid = (512/32, M/128), block = 256; each wave: one 16-row m-tile x two n-tiles.
__global__ __launch_bounds__(256) void gemm512_kernel(const _Float16* __restrict__ X,
        const _Float16* __restrict__ Wt, const float* __restrict__ bias,
        float* __restrict__ out32, _Float16* __restrict__ out16) {
    const int tid  = threadIdx.x;
    const int wave = tid >> 5, lane = tid & 31;
    const int lr   = lane & 15;
    const int hi   = (lane >> 4) & 1;
    const int hi8  = hi << 3, hi16 = hi << 4;
    const int n0   = blockIdx.x << 5;
    const int m0   = (blockIdx.y << 7) + (wave << 4);
    const _Float16* arow  = X  + (size_t)(m0 + lr) * DM;
    const _Float16* brow0 = Wt + (size_t)(n0 + lr) * DM;
    const _Float16* brow1 = Wt + (size_t)(n0 + 16 + lr) * DM;

    v8f acc0 = {}, acc1 = {};
    for (int k0 = 0; k0 < DM; k0 += 32) {
        v16h a  = load_a_contig(arow + k0 + hi16);
        v16h b0 = load_b_contig(brow0 + k0 + hi16);
        v16h b1 = load_b_contig(brow1 + k0 + hi16);
        acc0 = __builtin_amdgcn_wmma_f32_16x16x32_f16(false, a, false, b0, (short)0, acc0, false, false);
        acc1 = __builtin_amdgcn_wmma_f32_16x16x32_f16(false, a, false, b1, (short)0, acc1, false, false);
    }
    const float bb0 = bias[n0 + lr];
    const float bb1 = bias[n0 + 16 + lr];
#pragma unroll
    for (int r = 0; r < 8; ++r) {
        const size_t base = (size_t)(m0 + r + hi8) * DM;
        const float v0 = acc0[r] + bb0;
        const float v1 = acc1[r] + bb1;
        if (out32) { out32[base + n0 + lr] = v0; out32[base + n0 + 16 + lr] = v1; }
        if (out16) { out16[base + n0 + lr] = (_Float16)v0; out16[base + n0 + 16 + lr] = (_Float16)v1; }
    }
}

// ---------------------------------------- final projection: out = R + relu(X@W + b)
__global__ __launch_bounds__(256) void gemm_final_kernel(const _Float16* __restrict__ X,
        const _Float16* __restrict__ Wt, const float* __restrict__ bias,
        const float* __restrict__ resid, float* __restrict__ out) {
    const int tid  = threadIdx.x;
    const int wave = tid >> 5, lane = tid & 31;
    const int lr   = lane & 15;
    const int hi   = (lane >> 4) & 1;
    const int hi8  = hi << 3, hi16 = hi << 4;
    const int n0   = blockIdx.x << 5;
    const int m0   = (blockIdx.y << 7) + (wave << 4);
    const _Float16* arow  = X  + (size_t)(m0 + lr) * DM;
    const _Float16* brow0 = Wt + (size_t)(n0 + lr) * DM;
    const _Float16* brow1 = Wt + (size_t)(n0 + 16 + lr) * DM;

    v8f acc0 = {}, acc1 = {};
    for (int k0 = 0; k0 < DM; k0 += 32) {
        v16h a  = load_a_contig(arow + k0 + hi16);
        v16h b0 = load_b_contig(brow0 + k0 + hi16);
        v16h b1 = load_b_contig(brow1 + k0 + hi16);
        acc0 = __builtin_amdgcn_wmma_f32_16x16x32_f16(false, a, false, b0, (short)0, acc0, false, false);
        acc1 = __builtin_amdgcn_wmma_f32_16x16x32_f16(false, a, false, b1, (short)0, acc1, false, false);
    }
    const float bb0 = bias[n0 + lr];
    const float bb1 = bias[n0 + 16 + lr];
#pragma unroll
    for (int r = 0; r < 8; ++r) {
        const size_t base = (size_t)(m0 + r + hi8) * DM;
        float v0 = acc0[r] + bb0; v0 = v0 > 0.f ? v0 : 0.f;
        float v1 = acc1[r] + bb1; v1 = v1 > 0.f ? v1 : 0.f;
        out[base + n0 + lr]      = resid[base + n0 + lr] + v0;
        out[base + n0 + 16 + lr] = resid[base + n0 + 16 + lr] + v1;
    }
}

// ------------------------------------------------------------- attention
// One workgroup per (batch b, 16-row q tile), loops over all 8 heads so the
// head-mean attention map accumulates in registers (no atomics, no 256MB A).
// Vt is the per-batch transposed V: Vt[b][n(512)][key(1024)].
// grid = (NP/16, BATCH), block = 256 (8 waves)
__global__ __launch_bounds__(256) void attn_kernel(const _Float16* __restrict__ qh,
        const _Float16* __restrict__ kh, const _Float16* __restrict__ Vt,
        const float* __restrict__ q32, float* __restrict__ O32,
        float* __restrict__ Am) {
    __shared__ _Float16 sP[16][NP];      // 32KB: scores, then probabilities (f16)
    __shared__ float    sAux[4][16][16]; // 4KB: row reductions + cross-wave partials

    const int tid  = threadIdx.x;
    const int wave = tid >> 5, lane = tid & 31;
    const int lr   = lane & 15;
    const int hi   = (lane >> 4) & 1;
    const int hi8  = hi << 3, hi16 = hi << 4;
    const int q0   = blockIdx.x << 4;
    const int b    = blockIdx.y;
    const int row  = tid >> 4, sub = tid & 15;   // softmax mapping: 16 thr/row
    const float scale = 0.04419417382415922f;    // 1/sqrt(512)

    float amAcc[64];
#pragma unroll
    for (int j = 0; j < 64; ++j) amAcc[j] = 0.f;

    for (int h = 0; h < HEADS; ++h) {
        // q tile A-fragments (dh=64 -> two K=32 chunks), shared by all waves
        const _Float16* qrow = qh + ((size_t)(b * NP) + q0 + lr) * DM + h * DHD;
        v16h aq0 = load_a_contig(qrow + 0  + hi16);
        v16h aq1 = load_a_contig(qrow + 32 + hi16);

        // ---- S = q . k^T  (each wave does 8 of the 64 key tiles)
        for (int kt = wave; kt < 64; kt += 8) {
            const _Float16* krow = kh + ((size_t)(b * NP) + kt * 16 + lr) * DM + h * DHD;
            if (kt + 8 < 64)
                __builtin_prefetch(krow + (size_t)128 * DM, 0, 0);  // next kt row
            v16h b0 = load_b_contig(krow + 0  + hi16);
            v16h b1 = load_b_contig(krow + 32 + hi16);
            v8f c = {};
            c = __builtin_amdgcn_wmma_f32_16x16x32_f16(false, aq0, false, b0, (short)0, c, false, false);
            c = __builtin_amdgcn_wmma_f32_16x16x32_f16(false, aq1, false, b1, (short)0, c, false, false);
#pragma unroll
            for (int r = 0; r < 8; ++r)
                sP[r + hi8][kt * 16 + lr] = (_Float16)(c[r] * scale);
        }
        __syncthreads();

        // ---- softmax over 1024 keys, 16 threads per row, f32 stats
        float mx = -3.0e38f;
        for (int j = sub; j < NP; j += 16) mx = fmaxf(mx, (float)sP[row][j]);
        sAux[0][row][sub] = mx;
        __syncthreads();
#pragma unroll
        for (int i = 0; i < 16; ++i) mx = fmaxf(mx, sAux[0][row][i]);
        __syncthreads();
        float sum = 0.f;
        for (int j = sub; j < NP; j += 16) {
            float p = __expf((float)sP[row][j] - mx);
            sP[row][j] = (_Float16)p;
            sum += p;
        }
        sAux[0][row][sub] = sum;
        __syncthreads();
        float tot = 0.f;
#pragma unroll
        for (int i = 0; i < 16; ++i) tot += sAux[0][row][i];
        const float inv = 1.f / tot;
#pragma unroll
        for (int jj = 0; jj < 64; ++jj) {
            const int j = sub + (jj << 4);
            const float p = (float)sP[row][j] * inv;
            sP[row][j] = (_Float16)p;
            amAcc[jj] += p;     // head-mean accumulation (scaled by 1/8 at end)
        }
        __syncthreads();

        // ---- O = P @ V : 4 dh-tiles x 2 K-halves spread across 8 waves
        const int nt = wave & 3, khalf = wave >> 2;
        const _Float16* vrow = Vt + ((size_t)(b * DM) + h * DHD + nt * 16 + lr) * NP;
        v8f oc = {};
        for (int kk = khalf * 16; kk < khalf * 16 + 16; ++kk) {
            const int k0 = kk << 5;
            v16h ap  = load_a_contig(&sP[lr][k0 + hi16]);  // P from LDS
            v16h bvf = load_b_contig(vrow + k0 + hi16);    // key-contiguous V^T
            oc = __builtin_amdgcn_wmma_f32_16x16x32_f16(false, ap, false, bvf,
                                                        (short)0, oc, false, false);
        }
        if (wave >= 4) {
#pragma unroll
            for (int r = 0; r < 8; ++r) sAux[nt][r + hi8][lr] = oc[r];
        }
        __syncthreads();
        if (wave < 4) {
#pragma unroll
            for (int r = 0; r < 8; ++r) {
                const int m = r + hi8;
                const size_t idx = ((size_t)(b * NP) + q0 + m) * DM + h * DHD + nt * 16 + lr;
                O32[idx] = q32[idx] + oc[r] + sAux[nt][m][lr];  // residual inside head
            }
        }
        __syncthreads();
    }

    // mean attention over heads, written transposed: Am[b][k][q]
#pragma unroll
    for (int jj = 0; jj < 64; ++jj) {
        const int col = sub + (jj << 4);
        Am[((size_t)b * NP + col) * NP + q0 + row] = amAcc[jj] * 0.125f;
    }
}

// ---------------------------------------------------------------- launcher
extern "C" void kernel_launch(void* const* d_in, const int* in_sizes, int n_in,
                              void* d_out, int out_size, void* d_ws, size_t ws_size,
                              hipStream_t stream) {
    (void)in_sizes; (void)n_in; (void)out_size; (void)ws_size;
    const float* Q  = (const float*)d_in[0];
    const float* K  = (const float*)d_in[1];
    const float* Wq = (const float*)d_in[2];
    const float* bq = (const float*)d_in[3];
    const float* Wk = (const float*)d_in[4];
    const float* bk = (const float*)d_in[5];
    const float* Wv = (const float*)d_in[6];
    const float* bv = (const float*)d_in[7];
    const float* Wo = (const float*)d_in[8];
    const float* bo = (const float*)d_in[9];

    char* ws = (char*)d_ws;
    const size_t MB = 1024ull * 1024ull;
    _Float16* Qh   = (_Float16*)(ws + 0);                       // 8 MB
    _Float16* Kh   = (_Float16*)(ws + 8 * MB);                  // 8 MB
    _Float16* Wqt  = (_Float16*)(ws + 16 * MB);                 // 0.5 MB (transposed)
    _Float16* Wkt  = (_Float16*)(ws + 16 * MB + 512 * 1024);    // 0.5 MB
    _Float16* Wvt  = (_Float16*)(ws + 17 * MB);                 // 0.5 MB
    _Float16* Wot  = (_Float16*)(ws + 17 * MB + 512 * 1024);    // 0.5 MB
    float*    q32  = (float*)   (ws + 18 * MB);                 // 16 MB
    _Float16* qhp  = (_Float16*)(ws + 34 * MB);                 // 8 MB
    _Float16* khp  = (_Float16*)(ws + 42 * MB);                 // 8 MB
    _Float16* vhp  = (_Float16*)(ws + 50 * MB);                 // 8 MB
    _Float16* Vt   = (_Float16*)(ws + 58 * MB);                 // 8 MB (per-batch V^T)
    float*    O32  = (float*)   (ws + 66 * MB);                 // 16 MB
    _Float16* Oh16 = (_Float16*)(ws + 82 * MB);                 // 8 MB  (total 90 MB)

    const int nBN = BATCH * NP * DM;   // 4,194,304
    cvt_f32_to_f16<<<(nBN + 255) / 256, 256, 0, stream>>>(Q, Qh, nBN);
    cvt_f32_to_f16<<<(nBN + 255) / 256, 256, 0, stream>>>(K, Kh, nBN);

    // weights: f32 [k][n] -> f16 [n][k]
    dim3 tb(32, 8), tgW(DM / 32, DM / 32, 1);
    transpose_cvt_kernel<float><<<tgW, tb, 0, stream>>>(Wq, Wqt, DM, DM);
    transpose_cvt_kernel<float><<<tgW, tb, 0, stream>>>(Wk, Wkt, DM, DM);
    transpose_cvt_kernel<float><<<tgW, tb, 0, stream>>>(Wv, Wvt, DM, DM);
    transpose_cvt_kernel<float><<<tgW, tb, 0, stream>>>(Wo, Wot, DM, DM);

    dim3 gg(DM / 32, (BATCH * NP) / 128);   // (16, 64)
    gemm512_kernel<<<gg, 256, 0, stream>>>(Qh, Wqt, bq, q32, qhp);
    gemm512_kernel<<<gg, 256, 0, stream>>>(Kh, Wkt, bk, nullptr, khp);
    gemm512_kernel<<<gg, 256, 0, stream>>>(Kh, Wvt, bv, nullptr, vhp);

    // V: f16 [key(1024)][n(512)] -> f16 [n(512)][key(1024)], per batch
    dim3 tgV(DM / 32, NP / 32, BATCH);
    transpose_cvt_kernel<_Float16><<<tgV, tb, 0, stream>>>(vhp, Vt, NP, DM);

    float* AmOut = (float*)d_out + (size_t)BATCH * NP * DM;
    attn_kernel<<<dim3(NP / 16, BATCH), 256, 0, stream>>>(qhp, khp, Vt, q32, O32, AmOut);

    cvt_f32_to_f16<<<(nBN + 255) / 256, 256, 0, stream>>>(O32, Oh16, nBN);
    gemm_final_kernel<<<gg, 256, 0, stream>>>(Oh16, Wot, bo, O32, (float*)d_out);
}
